// LinearAttention_43593918054585
// MI455X (gfx1250) — matrix operand order repaired and checked by reading, not compiled, not run on hardware
//
#include <hip/hip_runtime.h>
#include <hip/hip_bf16.h>
#include <cstdint>
#include <cstddef>

// ---------------------------------------------------------------- types
typedef _Float16 f16;
typedef __attribute__((ext_vector_type(16))) _Float16 v16h;
typedef __attribute__((ext_vector_type(8)))  _Float16 h8;
typedef __attribute__((ext_vector_type(4)))  _Float16 h4;
typedef __attribute__((ext_vector_type(8)))  float    v8f;
typedef __attribute__((ext_vector_type(4)))  unsigned int u32x4;
typedef __attribute__((ext_vector_type(8)))  int i32x8;
typedef __attribute__((ext_vector_type(4)))  int i32x4;

union AFrag { v16h v; uint32_t u[8]; };
union BFrag { v16h v; h8 h[2]; };

// TDM available? (device pass on gfx1250; 6-arg builtin per this toolchain)
#if defined(__gfx1250__) && __has_builtin(__builtin_amdgcn_tensor_load_to_lds)
#define HAVE_TDM 1
#endif

// ---------------------------------------------------------------- shapes
constexpr int Bc  = 4;
constexpr int Tc  = 4096;
constexpr int Dc  = 2048;
constexpr int Hc  = 32;
constexpr int HDc = 64;
constexpr int MT  = Bc * Tc;        // 16384 rows total
constexpr int TSPLIT = 8;           // KV T-split for occupancy
constexpr float EPSf = 1e-6f;

// ================================================================ convert
__global__ __launch_bounds__(256)
void cvt_f32_f16(const float* __restrict__ s, f16* __restrict__ d, int n)
{
    int i = (blockIdx.x * 256 + threadIdx.x) * 4;
    if (i < n) {
        const float4 f = *reinterpret_cast<const float4*>(s + i);
        h4 o;
        o.x = (f16)f.x; o.y = (f16)f.y; o.z = (f16)f.z; o.w = (f16)f.w;
        *reinterpret_cast<h4*>(d + i) = o;
    }
}

// ================================================================ GEMM
// C[M,N] = act(A[M,K] * W[N,K]^T + bias)
// 256 threads = 8 waves in 2x4 grid; per-wave 64x32 = 4x2 WMMA tiles.
#define BM 128
#define BN 128
#define BK 32
#define LDA (BK + 8)   // 40 halves / row  -> 80B rows (16 dwords data + 4 dwords pad)
#define LDB (BN + 8)   // 136 halves / row -> 272B, 16B aligned rows

__global__ __launch_bounds__(256)
void gemm_f16(const f16* __restrict__ A, const f16* __restrict__ W,
              const float* __restrict__ bias,
              f16* __restrict__ Ch, float* __restrict__ Cf,
              int M, int N, int K, int act)
{
    __shared__ __align__(16) f16 As[2][BM][LDA];   // [row m][k]
    __shared__ __align__(16) f16 Bs[2][BK][LDB];   // transposed: [k][n]

    const int tid  = threadIdx.x;
    const int lane = tid & 31;
    const int w    = tid >> 5;
    const int wm   = w >> 2;            // 0..1
    const int wn   = w & 3;             // 0..3
    const int m0   = blockIdx.y * BM;
    const int n0   = blockIdx.x * BN;
    const int nk   = K / BK;

    v8f acc[4][2];
    #pragma unroll
    for (int i = 0; i < 4; ++i)
        #pragma unroll
        for (int j = 0; j < 2; ++j)
            acc[i][j] = (v8f){0.f,0.f,0.f,0.f,0.f,0.f,0.f,0.f};

#if defined(HAVE_TDM)
    // One wave issues a TDM descriptor: DMA the 128x32 f16 A-tile into LDS,
    // with D# padding reproducing the LDA=40-half row stride
    // (pad_interval: 16 dwords -> code 3, pad_amount: 4 dwords -> code 3).
    auto issue_tdm_a = [&](int kt, int buf) {
        if (tid < 32) {
            const uint64_t ga =
                (uint64_t)(uintptr_t)(A + (size_t)m0 * K + (size_t)kt * BK);
            const unsigned ldso = (unsigned)(uintptr_t)(&As[buf][0][0]);
            u32x4 g0 = { 1u,                                  // count=1, user desc
                         ldso,                                // lds_addr
                         (unsigned)ga,                        // global_addr[31:0]
                         (unsigned)(ga >> 32) | (2u << 30) }; // ga[56:32] | type=2
            i32x8 g1;
            g1[0] = (int)((1u << 16)      // data_size = 2 bytes
                        | (1u << 20)      // pad_enable
                        | (3u << 22)      // pad_interval = 16 dwords
                        | (3u << 25));    // pad_amount   = 4 dwords
            g1[1] = (int)((unsigned)K << 16);                         // tensor_dim0 lo
            g1[2] = (int)(((unsigned)K >> 16) | ((unsigned)M << 16)); // dim0 hi | dim1 lo
            g1[3] = (int)(((unsigned)M >> 16) | ((unsigned)BK << 16));// dim1 hi | tile_dim0
            g1[4] = BM;                                               // tile_dim1 (tile_dim2=0)
            g1[5] = K;                                                // tensor_dim0_stride lo
            g1[6] = 0;
            g1[7] = 0;
            i32x4 gz4 = { 0, 0, 0, 0 };
            i32x8 gz8 = { 0, 0, 0, 0, 0, 0, 0, 0 };
            __builtin_amdgcn_tensor_load_to_lds(g0, g1, gz4, gz4, gz8, 0);
        }
    };
#else
    auto load_a_tile = [&](int kt, int buf) {
        const int k0 = kt * BK;
        #pragma unroll
        for (int r = 0; r < 2; ++r) {
            const int slot = tid + r * 256;
            const int row  = slot >> 2;
            const int c8   = (slot & 3) * 8;
            h8 a = *reinterpret_cast<const h8*>(A + (size_t)(m0 + row) * K + k0 + c8);
            *reinterpret_cast<h8*>(&As[buf][row][c8]) = a;
        }
    };
#endif

    auto load_b_tile = [&](int kt, int buf) {
        const int k0 = kt * BK;
        #pragma unroll
        for (int r = 0; r < 2; ++r) {
            const int slot = tid + r * 256;      // 0..511
            const int row  = slot >> 2;          // 0..127 (n-row)
            const int c8   = (slot & 3) * 8;     // 0,8,16,24
            h8 bvec = *reinterpret_cast<const h8*>(W + (size_t)(n0 + row) * K + k0 + c8);
            #pragma unroll
            for (int j = 0; j < 8; ++j) Bs[buf][c8 + j][row] = bvec[j];
        }
        if (kt + 2 < nk) {   // lookahead prefetch -> global_prefetch_b8
            __builtin_prefetch(W + (size_t)(n0 + (tid >> 1)) * K + (size_t)(kt + 2) * BK, 0, 1);
        }
    };

#if defined(HAVE_TDM)
    issue_tdm_a(0, 0);
    load_b_tile(0, 0);
    __builtin_amdgcn_s_wait_tensorcnt(0);
#else
    load_a_tile(0, 0);
    load_b_tile(0, 0);
#endif
    __syncthreads();

    for (int kt = 0; kt < nk; ++kt) {
        const int cur = kt & 1;
        if (kt + 1 < nk) {
#if defined(HAVE_TDM)
            issue_tdm_a(kt + 1, cur ^ 1);
#else
            load_a_tile(kt + 1, cur ^ 1);
#endif
            load_b_tile(kt + 1, cur ^ 1);
        }

        // A fragments: lane m = lane&15, K interleave per ISA 16-bit A layout
        AFrag af[4];
        const int am = lane & 15;
        const int as = lane >> 4;
        #pragma unroll
        for (int im = 0; im < 4; ++im) {
            const int row = wm * 64 + im * 16 + am;
            #pragma unroll
            for (int v = 0; v < 8; ++v) {
                const int kk = (v & 3) * 2 + as * 8 + (v >> 2) * 16;
                af[im].u[v] = *reinterpret_cast<const uint32_t*>(&As[cur][row][kk]);
            }
        }
        // B fragments: lane n = lane&15, K interleave per ISA 16-bit layout (same as A)
        BFrag bf[2];
        #pragma unroll
        for (int in = 0; in < 2; ++in) {
            const int nb = wn * 32 + in * 16;
            #pragma unroll
            for (int s = 0; s < 16; ++s)
                bf[in].v[s] = Bs[cur][(s & 7) + (lane >> 4) * 8 + (s >> 3) * 16][nb + (lane & 15)];
        }
        #pragma unroll
        for (int im = 0; im < 4; ++im)
            #pragma unroll
            for (int in = 0; in < 2; ++in)
                acc[im][in] = __builtin_amdgcn_wmma_f32_16x16x32_f16(
                    false, af[im].v, false, bf[in].v,
                    (short)0, acc[im][in], false, false);

#if defined(HAVE_TDM)
        __builtin_amdgcn_s_wait_tensorcnt(0);  // DMA of next tile complete
#endif
        __syncthreads();
    }

    // epilogue: C layout row = i + 8*(lane>=16), col = lane&15
    const int cs = lane >> 4;
    const int cn = lane & 15;
    #pragma unroll
    for (int im = 0; im < 4; ++im) {
        #pragma unroll
        for (int in = 0; in < 2; ++in) {
            const int gn = n0 + wn * 32 + in * 16 + cn;
            const float bv = bias ? bias[gn] : 0.0f;
            #pragma unroll
            for (int i = 0; i < 8; ++i) {
                const int gm = m0 + wm * 64 + im * 16 + i + 8 * cs;
                float val = acc[im][in][i] + bv;
                if (act) val = (val > 0.0f) ? (val + 1.0f) : __expf(val); // elu(x)+1
                if (Ch) Ch[(size_t)gm * N + gn] = (f16)val;
                if (Cf) Cf[(size_t)gm * N + gn] = val;
            }
        }
    }
}

// ================================================================ Ksum
__global__ __launch_bounds__(256)
void ksum_kernel(const f16* __restrict__ Kh, float* __restrict__ Ksum)
{
    const int bh = blockIdx.x;
    const int b = bh / Hc, h = bh % Hc;
    const int dk = threadIdx.x & 63;
    const int part = threadIdx.x >> 6;     // 4-way T partition
    const f16* base = Kh + (size_t)b * Tc * Dc + (size_t)h * HDc + dk;
    float s = 0.0f;
    for (int t = part; t < Tc; t += 4) s += (float)base[(size_t)t * Dc];
    __shared__ float red[256];
    red[threadIdx.x] = s;
    __syncthreads();
    if (part == 0)
        Ksum[(size_t)bh * HDc + dk] = red[dk] + red[64 + dk] + red[128 + dk] + red[192 + dk];
}

// ================================================================ KV = K^T V  (partials over T-split)
#define KVLDK 40   // KT row pad (32 t + 8)
#define KVLDV 72   // Vt row pad (64 dv + 8)
__global__ __launch_bounds__(256)
void kv_partial(const f16* __restrict__ Kh, const f16* __restrict__ Vh,
                float* __restrict__ KVp)
{
    __shared__ __align__(16) f16 KT[HDc][KVLDK];  // transposed: [dk][t]
    __shared__ __align__(16) f16 Vt[32][KVLDV];   // [t][dv]

    const int bh = blockIdx.x;
    const int ts = blockIdx.y;
    const int b = bh / Hc, h = bh % Hc;
    const int tid = threadIdx.x, lane = tid & 31, w = tid >> 5;
    const int im = w >> 1;           // dk tile 0..3
    const int ip = w & 1;            // dv half 0..1

    v8f acc[2];
    acc[0] = (v8f){0.f,0.f,0.f,0.f,0.f,0.f,0.f,0.f};
    acc[1] = (v8f){0.f,0.f,0.f,0.f,0.f,0.f,0.f,0.f};

    const size_t baseG = (size_t)b * Tc * Dc + (size_t)h * HDc;
    const int tbase = ts * (Tc / TSPLIT);
    const int trow = tid >> 3;            // 0..31
    const int c8   = (tid & 7) * 8;       // 0..56

    for (int tc = 0; tc < (Tc / TSPLIT) / 32; ++tc) {
        const int t0 = tbase + tc * 32;
        h8 kv8 = *reinterpret_cast<const h8*>(Kh + baseG + (size_t)(t0 + trow) * Dc + c8);
        #pragma unroll
        for (int j = 0; j < 8; ++j) KT[c8 + j][trow] = kv8[j];
        h8 vv8 = *reinterpret_cast<const h8*>(Vh + baseG + (size_t)(t0 + trow) * Dc + c8);
        *reinterpret_cast<h8*>(&Vt[trow][c8]) = vv8;
        __syncthreads();

        AFrag af;
        const int dk = im * 16 + (lane & 15);
        const int as = lane >> 4;
        #pragma unroll
        for (int v = 0; v < 8; ++v) {
            const int kk = (v & 3) * 2 + as * 8 + (v >> 2) * 16;
            af.u[v] = *reinterpret_cast<const uint32_t*>(&KT[dk][kk]);
        }
        #pragma unroll
        for (int j = 0; j < 2; ++j) {
            BFrag bf;
            const int nb = ip * 32 + j * 16;
            #pragma unroll
            for (int s = 0; s < 16; ++s)
                bf.v[s] = Vt[(s & 7) + (lane >> 4) * 8 + (s >> 3) * 16][nb + (lane & 15)];
            acc[j] = __builtin_amdgcn_wmma_f32_16x16x32_f16(
                false, af.v, false, bf.v, (short)0, acc[j], false, false);
        }
        __syncthreads();
    }

    float* outp = KVp + ((size_t)bh * TSPLIT + ts) * (HDc * HDc);
    const int cs = lane >> 4, cn = lane & 15;
    #pragma unroll
    for (int j = 0; j < 2; ++j) {
        const int dv = ip * 32 + j * 16 + cn;
        #pragma unroll
        for (int i = 0; i < 8; ++i) {
            const int dk = im * 16 + i + 8 * cs;
            outp[dk * HDc + dv] = acc[j][i];
        }
    }
}

__global__ __launch_bounds__(256)
void kv_reduce(const float* __restrict__ KVp, f16* __restrict__ KVh)
{
    const int bh = blockIdx.x;
    for (int idx = threadIdx.x; idx < HDc * HDc; idx += 256) {
        float s = 0.0f;
        #pragma unroll
        for (int p = 0; p < TSPLIT; ++p)
            s += KVp[((size_t)bh * TSPLIT + p) * (HDc * HDc) + idx];
        KVh[(size_t)bh * (HDc * HDc) + idx] = (f16)s;
    }
}

// ================================================================ ctx = (Q @ KV) * Z
#define ZROWS 128
__global__ __launch_bounds__(256)
void zq_context(const f16* __restrict__ Qh, const f16* __restrict__ KVh,
                const float* __restrict__ Ksum, f16* __restrict__ Ctx)
{
    __shared__ __align__(16) f16 Qs[ZROWS][HDc + 8];
    __shared__ __align__(16) f16 Bkv[HDc][HDc + 8];
    __shared__ float zs[ZROWS];
    __shared__ float ks[HDc];

    const int r0 = blockIdx.x * ZROWS;
    const int h  = blockIdx.y;
    const int b  = r0 / Tc;
    const int bh = b * Hc + h;
    const int tid = threadIdx.x, lane = tid & 31, w = tid >> 5;

    if (tid < HDc) ks[tid] = Ksum[(size_t)bh * HDc + tid];
    #pragma unroll
    for (int r = 0; r < 2; ++r) {                 // KV: 512 h8 slots
        const int slot = tid + r * 256;
        const int dk = slot >> 3, c8 = (slot & 7) * 8;
        h8 t = *reinterpret_cast<const h8*>(KVh + (size_t)bh * HDc * HDc + dk * HDc + c8);
        *reinterpret_cast<h8*>(&Bkv[dk][c8]) = t;
    }
    #pragma unroll
    for (int r = 0; r < 4; ++r) {                 // Q tile: 1024 h8 slots
        const int slot = tid + r * 256;
        const int row = slot >> 3, c8 = (slot & 7) * 8;
        h8 t = *reinterpret_cast<const h8*>(Qh + (size_t)(r0 + row) * Dc + (size_t)h * HDc + c8);
        *reinterpret_cast<h8*>(&Qs[row][c8]) = t;
    }
    __syncthreads();

    if (tid < ZROWS) {
        float dot = 0.0f;
        #pragma unroll
        for (int k = 0; k < HDc; ++k) dot += (float)Qs[tid][k] * ks[k];
        zs[tid] = 1.0f / (dot + EPSf);
    }
    __syncthreads();

    v8f acc[4];
    #pragma unroll
    for (int i = 0; i < 4; ++i) acc[i] = (v8f){0.f,0.f,0.f,0.f,0.f,0.f,0.f,0.f};

    #pragma unroll
    for (int kstep = 0; kstep < 2; ++kstep) {
        AFrag af;
        const int row = w * 16 + (lane & 15);
        const int as = lane >> 4;
        #pragma unroll
        for (int v = 0; v < 8; ++v) {
            const int kk = kstep * 32 + (v & 3) * 2 + as * 8 + (v >> 2) * 16;
            af.u[v] = *reinterpret_cast<const uint32_t*>(&Qs[row][kk]);
        }
        #pragma unroll
        for (int in = 0; in < 4; ++in) {
            BFrag bf;
            #pragma unroll
            for (int s = 0; s < 16; ++s)
                bf.v[s] = Bkv[kstep * 32 + (s & 7) + (lane >> 4) * 8 + (s >> 3) * 16][in * 16 + (lane & 15)];
            acc[in] = __builtin_amdgcn_wmma_f32_16x16x32_f16(
                false, af.v, false, bf.v, (short)0, acc[in], false, false);
        }
    }

    const int cs = lane >> 4, cn = lane & 15;
    #pragma unroll
    for (int in = 0; in < 4; ++in) {
        #pragma unroll
        for (int i = 0; i < 8; ++i) {
            const int rr = w * 16 + i + 8 * cs;
            const float val = acc[in][i] * zs[rr];
            Ctx[(size_t)(r0 + rr) * Dc + (size_t)h * HDc + in * 16 + cn] = (f16)val;
        }
    }
}

// ================================================================ launch
extern "C" void kernel_launch(void* const* d_in, const int* in_sizes, int n_in,
                              void* d_out, int out_size, void* d_ws, size_t ws_size,
                              hipStream_t stream)
{
    (void)in_sizes; (void)n_in; (void)out_size; (void)ws_size;

    const float* x  = (const float*)d_in[0];
    const float* Wq = (const float*)d_in[1];
    const float* bq = (const float*)d_in[2];
    const float* Wk = (const float*)d_in[3];
    const float* bk = (const float*)d_in[4];
    const float* Wv = (const float*)d_in[5];
    const float* bv = (const float*)d_in[6];
    const float* Wo = (const float*)d_in[7];
    const float* bo = (const float*)d_in[8];
    float* out = (float*)d_out;

    char* base = (char*)d_ws;
    size_t off = 0;
    auto take = [&](size_t bytes) -> char* {
        char* q = base + off;
        off += (bytes + 255) & ~(size_t)255;
        return q;
    };

    f16* xb   = (f16*)take((size_t)MT * Dc * sizeof(f16));
    f16* wqb  = (f16*)take((size_t)Dc * Dc * sizeof(f16));
    f16* wkb  = (f16*)take((size_t)Dc * Dc * sizeof(f16));
    f16* wvb  = (f16*)take((size_t)Dc * Dc * sizeof(f16));
    f16* wob  = (f16*)take((size_t)Dc * Dc * sizeof(f16));
    f16* Qh   = (f16*)take((size_t)MT * Dc * sizeof(f16));
    f16* Kh   = (f16*)take((size_t)MT * Dc * sizeof(f16));
    f16* Vh   = (f16*)take((size_t)MT * Dc * sizeof(f16));
    float* Ks = (float*)take((size_t)Bc * Hc * HDc * sizeof(float));
    float* KVp = (float*)take((size_t)Bc * Hc * TSPLIT * HDc * HDc * sizeof(float));
    f16* KVh  = (f16*)take((size_t)Bc * Hc * HDc * HDc * sizeof(f16));
    f16* ctx  = xb;   // x no longer needed once Q/K/V exist

    const int nX = MT * Dc;       // 33,554,432
    const int nW = Dc * Dc;       // 4,194,304

    cvt_f32_f16<<<nX / 1024, 256, 0, stream>>>(x,  xb,  nX);
    cvt_f32_f16<<<nW / 1024, 256, 0, stream>>>(Wq, wqb, nW);
    cvt_f32_f16<<<nW / 1024, 256, 0, stream>>>(Wk, wkb, nW);
    cvt_f32_f16<<<nW / 1024, 256, 0, stream>>>(Wv, wvb, nW);
    cvt_f32_f16<<<nW / 1024, 256, 0, stream>>>(Wo, wob, nW);

    dim3 gg(Dc / BN, MT / BM);    // (16, 128)
    gemm_f16<<<gg, 256, 0, stream>>>(xb, wqb, bq, Qh, nullptr, MT, Dc, Dc, 1);
    gemm_f16<<<gg, 256, 0, stream>>>(xb, wkb, bk, Kh, nullptr, MT, Dc, Dc, 1);
    gemm_f16<<<gg, 256, 0, stream>>>(xb, wvb, bv, Vh, nullptr, MT, Dc, Dc, 0);

    ksum_kernel<<<Bc * Hc, 256, 0, stream>>>(Kh, Ks);
    kv_partial<<<dim3(Bc * Hc, TSPLIT), 256, 0, stream>>>(Kh, Vh, KVp);
    kv_reduce<<<Bc * Hc, 256, 0, stream>>>(KVp, KVh);

    zq_context<<<dim3(MT / ZROWS, Hc), 256, 0, stream>>>(Qh, KVh, Ks, ctx);

    gemm_f16<<<gg, 256, 0, stream>>>(ctx, wob, bo, nullptr, out, MT, Dc, Dc, 0);
}
